// KANLayer_12979391169009
// MI455X (gfx1250) — compile-verified, matrix operand order checked by reading, not run on hardware
//
#include <hip/hip_runtime.h>
#include <hip/hip_bf16.h>

typedef __attribute__((ext_vector_type(16))) _Float16 v16h;
typedef __attribute__((ext_vector_type(2)))  _Float16 v2h;
typedef __attribute__((ext_vector_type(2)))  __fp16   v2hh;   // cvt_pkrtz return type
typedef __attribute__((ext_vector_type(8)))  float    v8f;

#define D_IN  32
#define D_OUT 32
#define HDIM  8
#define KTOT  256   // D_IN * HDIM, contraction length per (n,j)

__device__ __forceinline__ float fast_tanh(float x) {
#if __has_builtin(__builtin_amdgcn_tanhf)
  return __builtin_amdgcn_tanhf(x);           // native v_tanh_f32 on gfx1250
#elif __has_builtin(__builtin_amdgcn_exp2f)
  // tanh(x) = (1 - e^{-2x}) / (1 + e^{-2x}), e^{-2x} = 2^{-2x*log2(e)}
  float e = __builtin_amdgcn_exp2f(-2.88539008177792681f * x);
  return (1.0f - e) * __builtin_amdgcn_rcpf(1.0f + e);
#else
  float e = __expf(-2.0f * x);
  return (1.0f - e) / (1.0f + e);
#endif
}

// two activations (tanh MLP with W2 folded in), packed to 2 x f16
__device__ __forceinline__ v2h act2(float x, float w0, float c0, float s0,
                                             float w1, float c1, float s1) {
  float a0 = fast_tanh(fmaf(x, w0, c0)) * s0;
  float a1 = fast_tanh(fmaf(x, w1, c1)) * s1;
  v2hh t = __builtin_amdgcn_cvt_pkrtz(a0, a1);
  return __builtin_bit_cast(v2h, t);
}

__global__ __launch_bounds__(256)
void kan_wmma_kernel(const float* __restrict__ x,
                     const float* __restrict__ W1,
                     const float* __restrict__ b1,
                     const float* __restrict__ W2,
                     const float* __restrict__ b2,
                     float* __restrict__ out) {
  const int lane = threadIdx.x & 31;
  const int m15  = lane & 15;            // A-row / D-column index within tile
  const int hi   = lane >> 4;            // which half of the wave
  const int gwave = (blockIdx.x * blockDim.x + threadIdx.x) >> 5;
  const int ntile = gwave >> 1;          // 16-row tile of n
  const int jt    = gwave & 1;           // 16-col tile of j

  const int   n_a  = ntile * 16 + m15;   // row this lane builds A-fragments for
  const float* xrow = x + n_a * D_IN;

  // sum_i b2[j,i] for this lane's output column j = jt*16 + m15
  const int jcol = jt * 16 + m15;
  float b2s = 0.f;
  {
    const float4* bp = (const float4*)(b2 + jcol * D_IN);
    #pragma unroll
    for (int q = 0; q < 8; ++q) {
      float4 v = bp[q];
      b2s += v.x + v.y + v.z + v.w;
    }
  }

  v8f acc = {};

  #pragma unroll 1
  for (int c = 0; c < 16; ++c) {         // j-column of the tile (B fixed per c)
    // B: one-hot column c is all ones (K-slot permutation invariant);
    // built ONCE per c, reused by the 8 K-chunk WMMAs below.
    const _Float16 m = (m15 == c) ? (_Float16)1.0f : (_Float16)0.0f;
    v16h Bf;
    #pragma unroll
    for (int e = 0; e < 16; ++e) Bf[e] = m;

    const int j = jt * 16 + c;
    const float* w1r = W1 + j * KTOT + hi * 8;
    const float* b1r = b1 + j * KTOT + hi * 8;
    const float* w2r = W2 + j * KTOT + hi * 8;

    #pragma unroll 2
    for (int kc = 0; kc < 8; ++kc) {     // K-chunks of 32 (4 i-values x 8 h)
      // A-fragment K layout: lane half 'hi', elements e<8 -> i = 4*kc+hi,
      // e>=8 -> i = 4*kc+2+hi
      const float xA = xrow[kc * 4 + hi];
      const float xB = xrow[kc * 4 + 2 + hi];

      const float4* w1p = (const float4*)(w1r + kc * 32);
      const float4* b1p = (const float4*)(b1r + kc * 32);
      const float4* w2p = (const float4*)(w2r + kc * 32);
      float4 wA = w1p[0], wB = w1p[1], wC = w1p[4], wD = w1p[5];
      float4 cA = b1p[0], cB = b1p[1], cC = b1p[4], cD = b1p[5];
      float4 sA = w2p[0], sB = w2p[1], sC = w2p[4], sD = w2p[5];

      union { v16h v; v2h p[8]; } A;
      A.p[0] = act2(xA, wA.x, cA.x, sA.x, wA.y, cA.y, sA.y);
      A.p[1] = act2(xA, wA.z, cA.z, sA.z, wA.w, cA.w, sA.w);
      A.p[2] = act2(xA, wB.x, cB.x, sB.x, wB.y, cB.y, sB.y);
      A.p[3] = act2(xA, wB.z, cB.z, sB.z, wB.w, cB.w, sB.w);
      A.p[4] = act2(xB, wC.x, cC.x, sC.x, wC.y, cC.y, sC.y);
      A.p[5] = act2(xB, wC.z, cC.z, sC.z, wC.w, cC.w, sC.w);
      A.p[6] = act2(xB, wD.x, cD.x, sD.x, wD.y, cD.y, sD.y);
      A.p[7] = act2(xB, wD.z, cD.z, sD.z, wD.w, cD.w, sD.w);

      // D[n, c] += sum_K  tanh(x*W1+b1)*W2   (other columns get += 0)
      acc = __builtin_amdgcn_wmma_f32_16x16x32_f16(
          false, A.v, false, Bf, (short)0, acc, false, false);
    }
  }

  // D layout: VGPR r holds row M = hi*8 + r, column N = m15
  #pragma unroll
  for (int r = 0; r < 8; ++r) {
    const int nrow = ntile * 16 + hi * 8 + r;
    out[nrow * D_OUT + jcol] = acc[r] + b2s;
  }
}

extern "C" void kernel_launch(void* const* d_in, const int* in_sizes, int n_in,
                              void* d_out, int out_size, void* d_ws, size_t ws_size,
                              hipStream_t stream) {
  const float* x  = (const float*)d_in[0];
  const float* W1 = (const float*)d_in[1];
  const float* b1 = (const float*)d_in[2];
  const float* W2 = (const float*)d_in[3];
  const float* b2 = (const float*)d_in[4];
  float* out = (float*)d_out;

  const int N = in_sizes[0] / D_IN;      // 16384
  // one wave per (16-n-tile, 16-j-tile): N/16 * 2 waves = N/8; 8 waves/block
  const int blocks = N / 64;             // 256 blocks of 256 threads
  kan_wmma_kernel<<<dim3(blocks), dim3(256), 0, stream>>>(x, W1, b1, W2, b2, out);
}
